// mGRU_76544907149447
// MI455X (gfx1250) — matrix operand order repaired and checked by reading, not compiled, not run on hardware
//
#include <hip/hip_runtime.h>
#include <math.h>

// GRU persistent-kernel implementation for MI455X (gfx1250, wave32, WMMA).
// x:[B,T,E] f32, h0:[B,H] f32, w_ih:[3H,E] f32, w_hh:[3H,H] f32 -> out:[B,T,H] f32.
// Strategy: bf16 WMMA (16x16x32) with f32 accumulation; x / w_ih / w_hh converted
// to bf16 once up front; the T=512 recurrence runs in a single persistent kernel
// with one device-wide barrier per step (atomic arrive + spin). Each wave owns a
// 16x16 output tile and all three gate columns (j, j+H, j+2H), so gates fuse
// in-register with no extra sync.

#define Bb 32
#define Tt 512
#define Ee 512
#define Hh 1024
#define NBLK 16      // persistent blocks, 8 waves each -> 128 wave-tasks

typedef __attribute__((ext_vector_type(16))) __bf16          bf16x16;
typedef __attribute__((ext_vector_type(8)))  unsigned short  u16x8;
typedef __attribute__((ext_vector_type(16))) unsigned short  u16x16;

__device__ __forceinline__ unsigned short f2bf(float f) {
    union { float f; unsigned u; } x; x.f = f;
    unsigned r = x.u + 0x7FFFu + ((x.u >> 16) & 1u);   // round-to-nearest-even
    return (unsigned short)(r >> 16);
}

__device__ __forceinline__ bf16x16 mk_frag(u16x8 lo, u16x8 hi) {
    u16x16 au = __builtin_shufflevector(lo, hi, 0,1,2,3,4,5,6,7,8,9,10,11,12,13,14,15);
    return __builtin_bit_cast(bf16x16, au);
}

__global__ void k_f2bf(const float* __restrict__ s, unsigned short* __restrict__ d, int n) {
    int i = blockIdx.x * blockDim.x + threadIdx.x;
    if (i < n) d[i] = f2bf(s[i]);
}

__global__ void k_init_h(const float* __restrict__ h0, float* __restrict__ hf,
                         unsigned short* __restrict__ hb, unsigned* __restrict__ bar) {
    int i = blockIdx.x * blockDim.x + threadIdx.x;
    if (i < Bb * Hh) { float v = h0[i]; hf[i] = v; hb[i] = f2bf(v); }
    if (i == 0) *bar = 0u;
}

__global__ void __launch_bounds__(256, 1)
k_gru_persistent(const unsigned short* __restrict__ xb,    // [B][T][E] bf16
                 const unsigned short* __restrict__ wih,   // [3H][E] bf16, row-major
                 const unsigned short* __restrict__ whh,   // [3H][H] bf16, row-major
                 float* __restrict__ hf,                   // [2][B][H] f32 ping-pong
                 unsigned short* __restrict__ hb,          // [2][B][H] bf16 ping-pong
                 float* __restrict__ out,                  // [B][T][H] f32
                 unsigned* __restrict__ bar)
{
    const int tid   = threadIdx.x;
    const int lane  = tid & 31;
    const int wave  = tid >> 5;
    const int wtask = blockIdx.x * 8 + wave;     // 0..127
    const int m0    = (wtask & 1) * 16;          // batch-row tile base (0 or 16)
    const int c0    = (wtask >> 1) * 16;         // hidden-column tile base (0..1008)

    const int lrow = lane & 15;
    const int lhi  = lane >> 4;                  // 0 or 1

    const int arow = m0 + lrow;                  // A row this lane loads
    // B fragment rows (weight rows) per gate; each lane reads 16 contiguous K
    // at kb + lhi*16 (contiguous order chosen here, same as A: lanes 0-15 hold K 0..15, lanes 16-31 K 16..31).
    const unsigned short* wihR = wih + (size_t)(c0 + lrow) * Ee;
    const unsigned short* wihZ = wih + (size_t)(Hh   + c0 + lrow) * Ee;
    const unsigned short* wihN = wih + (size_t)(2*Hh + c0 + lrow) * Ee;
    const unsigned short* whhR = whh + (size_t)(c0 + lrow) * Hh;
    const unsigned short* whhZ = whh + (size_t)(Hh   + c0 + lrow) * Hh;
    const unsigned short* whhN = whh + (size_t)(2*Hh + c0 + lrow) * Hh;

    const unsigned short* xrow = xb + ((size_t)arow * Tt) * Ee;   // + t*Ee per step

    for (int t = 0; t < Tt; ++t) {
        const int cur = t & 1, nxt = cur ^ 1;
        const float*          hfc = hf + (size_t)cur * (Bb * Hh);
        const unsigned short* hbc = hb + (size_t)cur * (Bb * Hh);

        typedef __attribute__((ext_vector_type(8))) float f32x8;
        f32x8 accR = {}, accZ = {}, accNi = {}, accNh = {};

        // ---- x_t @ w_ih^T  (K = E = 512) ----
        {
            const unsigned short* xa = xrow + (size_t)t * Ee;
            #pragma unroll 2
            for (int kb = 0; kb < Ee; kb += 32) {
                // A layout: lane holds K [kb+lhi*16 .. +15] (contiguous, matches B)
                bf16x16 a  = mk_frag(*(const u16x8*)(xa + kb + lhi * 16),
                                     *(const u16x8*)(xa + kb + lhi * 16 + 8));
                bf16x16 bR = __builtin_bit_cast(bf16x16, *(const u16x16*)(wihR + kb + lhi * 16));
                bf16x16 bZ = __builtin_bit_cast(bf16x16, *(const u16x16*)(wihZ + kb + lhi * 16));
                bf16x16 bN = __builtin_bit_cast(bf16x16, *(const u16x16*)(wihN + kb + lhi * 16));
                accR  = __builtin_amdgcn_wmma_f32_16x16x32_bf16(false, a, false, bR, (short)0, accR,  false, false);
                accZ  = __builtin_amdgcn_wmma_f32_16x16x32_bf16(false, a, false, bZ, (short)0, accZ,  false, false);
                accNi = __builtin_amdgcn_wmma_f32_16x16x32_bf16(false, a, false, bN, (short)0, accNi, false, false);
            }
        }

        // ---- h @ w_hh^T  (K = H = 1024) ----
        {
            const unsigned short* ha = hbc + (size_t)arow * Hh;
            #pragma unroll 2
            for (int kb = 0; kb < Hh; kb += 32) {
                bf16x16 a  = mk_frag(*(const u16x8*)(ha + kb + lhi * 16),
                                     *(const u16x8*)(ha + kb + lhi * 16 + 8));
                bf16x16 bR = __builtin_bit_cast(bf16x16, *(const u16x16*)(whhR + kb + lhi * 16));
                bf16x16 bZ = __builtin_bit_cast(bf16x16, *(const u16x16*)(whhZ + kb + lhi * 16));
                bf16x16 bN = __builtin_bit_cast(bf16x16, *(const u16x16*)(whhN + kb + lhi * 16));
                accR  = __builtin_amdgcn_wmma_f32_16x16x32_bf16(false, a, false, bR, (short)0, accR,  false, false);
                accZ  = __builtin_amdgcn_wmma_f32_16x16x32_bf16(false, a, false, bZ, (short)0, accZ,  false, false);
                accNh = __builtin_amdgcn_wmma_f32_16x16x32_bf16(false, a, false, bN, (short)0, accNh, false, false);
            }
        }

        // ---- gates + state update (C/D layout: VGPR v, lane -> (m0+v+8*lhi, c0+(lane&15))) ----
        {
            float*          hfn = hf + (size_t)nxt * (Bb * Hh);
            unsigned short* hbn = hb + (size_t)nxt * (Bb * Hh);
            const int col = c0 + lrow;
            #pragma unroll
            for (int v = 0; v < 8; ++v) {
                const int row = m0 + v + lhi * 8;
                float r = 1.0f / (1.0f + __expf(-accR[v]));
                float z = 1.0f / (1.0f + __expf(-accZ[v]));
                float n = tanhf(accNi[v] + r * accNh[v]);
                float hold = hfc[(size_t)row * Hh + col];
                float hnew = (1.0f - z) * n + z * hold;
                hfn[(size_t)row * Hh + col] = hnew;
                hbn[(size_t)row * Hh + col] = f2bf(hnew);
                out[((size_t)row * Tt + t) * Hh + col] = hnew;
            }
        }

        // ---- device-wide barrier: monotonic counter, one arrival per block ----
        __syncthreads();
        if (tid == 0) {
            __threadfence();
            __hip_atomic_fetch_add(bar, 1u, __ATOMIC_RELEASE, __HIP_MEMORY_SCOPE_AGENT);
            const unsigned target = (unsigned)NBLK * (unsigned)(t + 1);
            while (__hip_atomic_load(bar, __ATOMIC_ACQUIRE, __HIP_MEMORY_SCOPE_AGENT) < target)
                __builtin_amdgcn_s_sleep(2);
        }
        __syncthreads();
    }
}

extern "C" void kernel_launch(void* const* d_in, const int* in_sizes, int n_in,
                              void* d_out, int out_size, void* d_ws, size_t ws_size,
                              hipStream_t stream) {
    (void)in_sizes; (void)n_in; (void)out_size; (void)ws_size;
    const float* x    = (const float*)d_in[0];   // [32,512,512]
    const float* h0   = (const float*)d_in[1];   // [32,1024]
    const float* w_ih = (const float*)d_in[2];   // [3072,512]
    const float* w_hh = (const float*)d_in[3];   // [3072,1024]
    float* out = (float*)d_out;                  // [32,512,1024]

    char* ws = (char*)d_ws;
    const size_t nWih = (size_t)3 * Hh * Ee;     // 1,572,864
    const size_t nWhh = (size_t)3 * Hh * Hh;     // 3,145,728
    const size_t nX   = (size_t)Bb * Tt * Ee;    // 8,388,608
    unsigned short* wih_b = (unsigned short*)(ws);                 // 3,145,728 B
    unsigned short* whh_b = (unsigned short*)(ws + 3145728);       // 6,291,456 B
    unsigned short* x_b   = (unsigned short*)(ws + 9437184);       // 16,777,216 B
    float*          hf    = (float*)(ws + 26214400);               // 262,144 B
    unsigned short* hb    = (unsigned short*)(ws + 26476544);      // 131,072 B
    unsigned*       bar   = (unsigned*)(ws + 26607616);

    k_f2bf<<<(int)((nWih + 255) / 256), 256, 0, stream>>>(w_ih, wih_b, (int)nWih);
    k_f2bf<<<(int)((nWhh + 255) / 256), 256, 0, stream>>>(w_hh, whh_b, (int)nWhh);
    k_f2bf<<<(int)((nX   + 255) / 256), 256, 0, stream>>>(x,    x_b,   (int)nX);
    k_init_h<<<(Bb * Hh + 255) / 256, 256, 0, stream>>>(h0, hf, hb, bar);
    k_gru_persistent<<<NBLK, 256, 0, stream>>>(x_b, wih_b, whh_b, hf, hb, out, bar);
}
